// PyGNN_43542378447174
// MI455X (gfx1250) — matrix operand order repaired and checked by reading, not compiled, not run on hardware
//
#include <hip/hip_runtime.h>
#include <hip/hip_bf16.h>
#include <math.h>

// MI455X / gfx1250 implementation of the 4-band spectral GNN.
// Dense GEMMs -> v_wmma_f32_16x16x32_f16 (wave32, 16x16 tiles, f32 accum).
// SpMM (44 propagate passes, the memory-bound bulk) -> coalesced gather +
// global_atomic_add_f32 scatter, self-loop terms folded into the init kernel.

typedef __attribute__((ext_vector_type(16))) _Float16 v16h;
typedef __attribute__((ext_vector_type(8)))  float    v8f;

// ----------------------------- elementwise ---------------------------------

__global__ void k_fill(float* __restrict__ p, float v, int n) {
  int i = blockIdx.x * blockDim.x + threadIdx.x;
  if (i < n) p[i] = v;
}

__global__ void k_deg_count(float* __restrict__ deg, const int* __restrict__ idx, int n) {
  int i = blockIdx.x * blockDim.x + threadIdx.x;
  if (i < n) atomicAdd(&deg[idx[i]], 1.0f);
}

__global__ void k_rsqrt(float* __restrict__ p, int n) {
  int i = blockIdx.x * blockDim.x + threadIdx.x;
  if (i < n) p[i] = rsqrtf(p[i]);
}

// Xn = coef(i) * X ; mode 0: coef = dinv[i]^2 (gcn_norm self loop)
//                    mode 1: coef = 1 - dinvL[i]^2 (Laplacian diag: +I - norm self loop)
__global__ void k_spmm_init(float* __restrict__ Xn, const float* __restrict__ X,
                            const float* __restrict__ dinv, int mode, int total) {
  int i = blockIdx.x * blockDim.x + threadIdx.x;
  if (i >= total) return;
  float d = dinv[i >> 6];
  float c = (mode == 0) ? d * d : (1.0f - d * d);
  Xn[i] = c * X[i];
}

// Xn[col] += sgn * dinv[row]*dinv[col] * X[row]; 16 threads/edge, 4 floats each:
// consecutive threads touch consecutive features -> coalesced gather + atomics.
__global__ void k_spmm_edges(float* __restrict__ Xn, const float* __restrict__ X,
                             const int* __restrict__ rows, const int* __restrict__ cols,
                             const float* __restrict__ dinv, float sgn, int E) {
  int t = blockIdx.x * blockDim.x + threadIdx.x;
  int e = t >> 4;
  if (e >= E) return;
  int f = (t & 15) << 2;
  int r = rows[e], c = cols[e];
  float w = sgn * dinv[r] * dinv[c];
  const float4 xv = *(const float4*)(X + (size_t)r * 64 + f);
  float* dst = Xn + (size_t)c * 64 + f;
  atomicAdd(dst + 0, w * xv.x);
  atomicAdd(dst + 1, w * xv.y);
  atomicAdd(dst + 2, w * xv.z);
  atomicAdd(dst + 3, w * xv.w);
}

// out = (1-alpha)*xn + alpha*h  (alpha=0 -> copy)
__global__ void k_mix(float* __restrict__ out, const float* __restrict__ xn,
                      const float* __restrict__ h, float alpha, int total) {
  int i = blockIdx.x * blockDim.x + threadIdx.x;
  if (i < total) out[i] = (1.0f - alpha) * xn[i] + alpha * h[i];
}

__global__ void k_bias_relu(float* __restrict__ out, const float* __restrict__ in,
                            const float* __restrict__ bias, int total) {
  int i = blockIdx.x * blockDim.x + threadIdx.x;
  if (i < total) out[i] = fmaxf(in[i] + bias[i & 63], 0.0f);
}

// ------------------------------- WMMA GEMM ---------------------------------
// f16 A-fragment, 16x32 (MxK), documented CDNA5 layout:
//   lane L: M = L&15, half = L>>4;  elem pair (2i,2i+1) at K = (i&4?16:0)+half*8+(i&3)*2
__device__ __forceinline__ v16h load_a_frag(const float* __restrict__ A, int m, int kb, int half) {
  v16h a;
  const float* row = A + (size_t)m * 64;
#pragma unroll
  for (int i = 0; i < 8; ++i) {
    int k = kb + ((i & 4) ? 16 : 0) + half * 8 + ((i & 3) << 1);
    float2 f = *(const float2*)(row + k);
    a[2 * i]     = (_Float16)f.x;
    a[2 * i + 1] = (_Float16)f.y;
  }
  return a;
}

// B fragments pre-packed as [frag][plane 0..7][lane 0..31] dwords -> coalesced loads.
__device__ __forceinline__ v16h load_b_frag(const unsigned* __restrict__ frag, int lane) {
  union { v16h h; unsigned u[8]; } b;
#pragma unroll
  for (int i = 0; i < 8; ++i) b.u[i] = frag[i * 32 + lane];
  return b.h;
}

// Pack one weight matrix [Kdim x 64] (fp32 row-major) into f16 B fragments.
// B layout 32x16 (KxN): lane L: N = L&15, half = L>>4; VGPR i holds K = (i&4?16:0)+half*8+(i&3)*2 (+1).
// One wave per fragment; frag = kt*4 + nt.
__global__ void k_prep_bfrag(const float* __restrict__ W, unsigned* __restrict__ arena) {
  int frag = blockIdx.x;
  int kt = frag >> 2, nt = frag & 3;
  int lane = threadIdx.x;
  int half = lane >> 4;
  int n = nt * 16 + (lane & 15);
  union { v16h h; unsigned u[8]; } b;
#pragma unroll
  for (int i = 0; i < 8; ++i) {
    int k = kt * 32 + ((i & 4) ? 16 : 0) + half * 8 + ((i & 3) << 1);
    b.h[2 * i]     = (_Float16)W[(size_t)k * 64 + n];
    b.h[2 * i + 1] = (_Float16)W[(size_t)(k + 1) * 64 + n];
  }
#pragma unroll
  for (int i = 0; i < 8; ++i)
    arena[(size_t)frag * 256 + i * 32 + lane] = b.u[i];
}

// C = act(concat(A0,A1)[M x 32*ktiles] @ W[.. x 64] + bias). One wave -> 16x64 strip.
// act: 0 none, 1 relu, 2 tanh. Wave-uniform exit keeps EXEC all-ones for WMMA.
__global__ void k_wmma_gemm(const float* __restrict__ A0, const float* __restrict__ A1,
                            const unsigned* __restrict__ Bfrag,
                            const float* __restrict__ bias, float* __restrict__ C,
                            int Mtiles, int ktiles, int act) {
  int wave = threadIdx.x >> 5, lane = threadIdx.x & 31, half = lane >> 4;
  int mtile = blockIdx.x * 8 + wave;
  if (mtile >= Mtiles) return;
  int m = mtile * 16 + (lane & 15);
  v8f acc[4] = {};
  for (int kt = 0; kt < ktiles; ++kt) {
    int kbase = kt * 32;
    const float* Asrc = (kbase < 64) ? A0 : A1;   // concat([.,64],[.,64]) along K
    v16h a = load_a_frag(Asrc, m, kbase & 63, half);
#pragma unroll
    for (int nt = 0; nt < 4; ++nt) {
      v16h b = load_b_frag(Bfrag + (size_t)(kt * 4 + nt) * 256, lane);
      acc[nt] = __builtin_amdgcn_wmma_f32_16x16x32_f16(false, a, false, b,
                                                       (short)0, acc[nt], false, false);
    }
  }
#pragma unroll
  for (int nt = 0; nt < 4; ++nt) {
#pragma unroll
    for (int v = 0; v < 8; ++v) {
      int row = mtile * 16 + v + 8 * half;   // f32 C/D layout
      int n = nt * 16 + (lane & 15);
      float val = acc[nt][v];
      if (bias) val += bias[n];
      if (act == 1) val = fmaxf(val, 0.0f);
      else if (act == 2) val = tanhf(val);
      C[(size_t)row * 64 + n] = val;
    }
  }
}

// gate = sigmoid(xl@Wl + xh@Wh + bl + bh); out = gate*xl + (1-gate)*xh
__global__ void k_wmma_gate(const float* __restrict__ xl, const float* __restrict__ xh,
                            const unsigned* __restrict__ BfL, const unsigned* __restrict__ BfH,
                            const float* __restrict__ bl, const float* __restrict__ bh,
                            float* __restrict__ out, int Mtiles) {
  int wave = threadIdx.x >> 5, lane = threadIdx.x & 31, half = lane >> 4;
  int mtile = blockIdx.x * 8 + wave;
  if (mtile >= Mtiles) return;
  int m = mtile * 16 + (lane & 15);
  v8f acc[4] = {};
#pragma unroll
  for (int kt = 0; kt < 2; ++kt) {
    v16h a = load_a_frag(xl, m, kt * 32, half);
#pragma unroll
    for (int nt = 0; nt < 4; ++nt) {
      v16h b = load_b_frag(BfL + (size_t)(kt * 4 + nt) * 256, lane);
      acc[nt] = __builtin_amdgcn_wmma_f32_16x16x32_f16(false, a, false, b,
                                                       (short)0, acc[nt], false, false);
    }
  }
#pragma unroll
  for (int kt = 0; kt < 2; ++kt) {
    v16h a = load_a_frag(xh, m, kt * 32, half);
#pragma unroll
    for (int nt = 0; nt < 4; ++nt) {
      v16h b = load_b_frag(BfH + (size_t)(kt * 4 + nt) * 256, lane);
      acc[nt] = __builtin_amdgcn_wmma_f32_16x16x32_f16(false, a, false, b,
                                                       (short)0, acc[nt], false, false);
    }
  }
#pragma unroll
  for (int nt = 0; nt < 4; ++nt) {
#pragma unroll
    for (int v = 0; v < 8; ++v) {
      int row = mtile * 16 + v + 8 * half;
      int n = nt * 16 + (lane & 15);
      float p = acc[nt][v] + bl[n] + bh[n];
      float g = 1.0f / (1.0f + expf(-p));
      float a_ = xl[(size_t)row * 64 + n];
      float b_ = xh[(size_t)row * 64 + n];
      out[(size_t)row * 64 + n] = g * a_ + (1.0f - g) * b_;
    }
  }
}

// logits = x@W_out[64x40] + b_out ; log_softmax over 40. One wave per node.
__global__ void k_out_logsoftmax(const float* __restrict__ X, const float* __restrict__ W,
                                 const float* __restrict__ bias, float* __restrict__ out, int N) {
  int wave = threadIdx.x >> 5, lane = threadIdx.x & 31;
  int node = blockIdx.x * 8 + wave;
  if (node >= N) return;
  const float* x = X + (size_t)node * 64;
  int c0 = lane, c1 = lane + 32;
  bool v1 = (c1 < 40);
  float l0 = bias[c0], l1 = v1 ? bias[c1] : 0.0f;
  for (int k = 0; k < 64; ++k) {
    float xk = x[k];
    l0 += xk * W[k * 40 + c0];
    if (v1) l1 += xk * W[k * 40 + c1];
  }
  float a1 = v1 ? l1 : -INFINITY;
  float mx = fmaxf(l0, a1);
  for (int off = 16; off > 0; off >>= 1) mx = fmaxf(mx, __shfl_xor(mx, off, 32));
  float e = expf(l0 - mx) + (v1 ? expf(a1 - mx) : 0.0f);
  for (int off = 16; off > 0; off >>= 1) e += __shfl_xor(e, off, 32);
  float lse = logf(e);
  out[(size_t)node * 40 + c0] = l0 - mx - lse;
  if (v1) out[(size_t)node * 40 + c1] = l1 - mx - lse;
}

// --------------------------------- driver ----------------------------------

extern "C" void kernel_launch(void* const* d_in, const int* in_sizes, int n_in,
                              void* d_out, int out_size, void* d_ws, size_t ws_size,
                              hipStream_t stream) {
  const float* feature    = (const float*)d_in[0];
  const int*   edge_index = (const int*)d_in[1];
  const int*   sub_edges  = (const int*)d_in[2];
  const float* conv_W     = (const float*)d_in[3];
  const float* conv_b     = (const float*)d_in[4];
  const float* W_low_t    = (const float*)d_in[5];
  const float* b_low_t    = (const float*)d_in[6];
  const float* W_high_t   = (const float*)d_in[7];
  const float* b_high_t   = (const float*)d_in[8];
  const float* W_low      = (const float*)d_in[9];
  const float* b_low      = (const float*)d_in[10];
  const float* W_high     = (const float*)d_in[11];
  const float* b_high     = (const float*)d_in[12];
  const float* W_out      = (const float*)d_in[13];
  const float* b_out      = (const float*)d_in[14];

  const int N  = in_sizes[0] / 64;          // 100000 (divisible by 16)
  const int E  = in_sizes[1] / 2;           // 1600000
  const int NF = N * 64;
  const int Mtiles = N / 16;
  const int K = 10;

  // ---- workspace carve-out (floats) ----
  float* wsf       = (float*)d_ws;
  float* dinv_band = wsf;                       // 4N  (gcn_norm per band, col-deg+1)
  float* dinv_ei   = dinv_band + 4 * (size_t)N; // N   (gcn_norm on edge_index)
  float* dinvL     = dinv_ei + (size_t)N;       // 2N  (Laplacian row-deg+1, bands 2,3)
  float* T0 = dinvL + 2 * (size_t)N;            // NF  ping
  float* T1 = T0 + (size_t)NF;                  // NF  pong
  float* H  = T1 + (size_t)NF;                  // NF  APPNP anchor / final x
  float* Bb = H + (size_t)NF;                   // 4*NF band outputs
  unsigned* arena    = (unsigned*)(Bb + 4 * (size_t)NF);
  unsigned* fragConv = arena;                   // 4 * 8 * 256 dwords
  unsigned* fragWlt  = arena + 4 * 2048;        // 16 * 256
  unsigned* fragWht  = fragWlt + 4096;          // 16 * 256
  unsigned* fragWl   = fragWht + 4096;          // 8 * 256
  unsigned* fragWh   = fragWl + 2048;           // 8 * 256

  dim3 blk(256);
  int gN7  = (7 * N + 255) / 256;
  int gE   = (E + 255) / 256;
  int gE16 = (int)(((size_t)E * 16 + 255) / 256);
  int gNF  = (NF + 255) / 256;
  int gMM  = (Mtiles + 7) / 8;
  int gOUT = (N + 7) / 8;

  // ---- degrees -> dinv (all 7N contiguous) ----
  k_fill<<<gN7, blk, 0, stream>>>(dinv_band, 1.0f, 7 * N);   // self-loop baseline
  for (int b = 0; b < 4; ++b)                                // gcn_norm: deg over col
    k_deg_count<<<gE, blk, 0, stream>>>(dinv_band + (size_t)b * N,
                                        sub_edges + (size_t)b * 2 * E + E, E);
  k_deg_count<<<gE, blk, 0, stream>>>(dinv_ei, edge_index + E, E);
  for (int j = 0; j < 2; ++j)                                // laplacian: deg over row
    k_deg_count<<<gE, blk, 0, stream>>>(dinvL + (size_t)j * N,
                                        sub_edges + (size_t)(2 + j) * 2 * E, E);
  k_rsqrt<<<gN7, blk, 0, stream>>>(dinv_band, 7 * N);

  // ---- pack weights into WMMA B-fragments ----
  for (int b = 0; b < 4; ++b)
    k_prep_bfrag<<<8, 32, 0, stream>>>(conv_W + (size_t)b * 64 * 64,
                                       fragConv + (size_t)b * 2048);
  k_prep_bfrag<<<16, 32, 0, stream>>>(W_low_t,  fragWlt);
  k_prep_bfrag<<<16, 32, 0, stream>>>(W_high_t, fragWht);
  k_prep_bfrag<<<8, 32, 0, stream>>>(W_low,  fragWl);
  k_prep_bfrag<<<8, 32, 0, stream>>>(W_high, fragWh);

  // ---- per-band pipeline ----
  for (int b = 0; b < 4; ++b) {
    const int*   r_b  = sub_edges + (size_t)b * 2 * E;
    const int*   c_b  = r_b + E;
    const float* dv   = dinv_band + (size_t)b * N;
    float*       Bout = Bb + (size_t)b * NF;

    // T0 = feature @ conv_W[b]
    k_wmma_gemm<<<gMM, blk, 0, stream>>>(feature, nullptr,
                                         fragConv + (size_t)b * 2048,
                                         nullptr, T0, Mtiles, 2, 0);
    // T1 = A_hat(band) @ T0 ; T0 = relu(T1 + conv_b[b])
    k_spmm_init<<<gNF, blk, 0, stream>>>(T1, T0, dv, 0, NF);
    k_spmm_edges<<<gE16, blk, 0, stream>>>(T1, T0, r_b, c_b, dv, 1.0f, E);
    k_bias_relu<<<gNF, blk, 0, stream>>>(T0, T1, conv_b + (size_t)b * 64, NF);

    if (b < 2) {
      // APPNP-style low-pass on the full graph's gcn_norm
      k_mix<<<gNF, blk, 0, stream>>>(H, T0, T0, 0.0f, NF);   // H = x
      for (int k = 0; k < K; ++k) {
        k_spmm_init<<<gNF, blk, 0, stream>>>(T1, T0, dinv_ei, 0, NF);
        k_spmm_edges<<<gE16, blk, 0, stream>>>(T1, T0, edge_index, edge_index + E,
                                               dinv_ei, 1.0f, E);
        float alpha = (k < K - 1) ? 0.1f : 0.0f;
        float* dst = (k == K - 1) ? Bout : T0;
        k_mix<<<gNF, blk, 0, stream>>>(dst, T1, H, alpha, NF);
      }
    } else {
      // High-pass: x <- L_sym @ x = (1 - dinvL^2) x  -  dinvL[r]dinvL[c] x[r] over edges
      const float* dL = dinvL + (size_t)(b - 2) * N;
      for (int k = 0; k < K; ++k) {
        k_spmm_init<<<gNF, blk, 0, stream>>>(T1, T0, dL, 1, NF);
        k_spmm_edges<<<gE16, blk, 0, stream>>>(T1, T0, r_b, c_b, dL, -1.0f, E);
        float* dst = (k == K - 1) ? Bout : T0;
        k_mix<<<gNF, blk, 0, stream>>>(dst, T1, T1, 0.0f, NF);
      }
    }
  }

  // ---- fusion head ----
  // x_low_t = tanh(concat(B0,B1) @ W_low_t + b_low_t)
  k_wmma_gemm<<<gMM, blk, 0, stream>>>(Bb, Bb + (size_t)NF, fragWlt,
                                       b_low_t, T0, Mtiles, 4, 2);
  // x_high_t = relu(concat(B2,B3) @ W_high_t + b_high_t)
  k_wmma_gemm<<<gMM, blk, 0, stream>>>(Bb + 2 * (size_t)NF, Bb + 3 * (size_t)NF, fragWht,
                                       b_high_t, T1, Mtiles, 4, 1);
  // gated blend -> H
  k_wmma_gate<<<gMM, blk, 0, stream>>>(T0, T1, fragWl, fragWh, b_low, b_high, H, Mtiles);
  // logits + log_softmax -> d_out
  k_out_logsoftmax<<<gOUT, blk, 0, stream>>>(H, W_out, b_out, (float*)d_out, N);
}
